// DualAttentionRNN_34780645163406
// MI455X (gfx1250) — matrix operand order repaired and checked by reading, not compiled, not run on hardware
//
#include <hip/hip_runtime.h>
#include <math.h>

#define DEV_INLINE __device__ __forceinline__

constexpr int Bn = 256, Tn = 64, Fn = 256, Hn = 64;

typedef __attribute__((ext_vector_type(16))) _Float16 v16h;
typedef __attribute__((ext_vector_type(8)))  _Float16 v8h;
typedef __attribute__((ext_vector_type(8)))  float    v8f;

DEV_INLINE float d_tanh(float x) {
#if __has_builtin(__builtin_amdgcn_tanhf)
  return __builtin_amdgcn_tanhf(x);
#else
  return tanhf(x);
#endif
}
DEV_INLINE float d_sigm(float x) { return 1.0f / (1.0f + __expf(-x)); }

// A fragment: 16(M) x 32(K) f16, row-major in LDS, row stride lda (halfs).
// Register element j <-> K = 8*hv + j (j<8), K = 16 + 8*hv + (j-8) (j>=8):
// two contiguous 16-byte chunks per lane -> 2x ds_load_b128.
DEV_INLINE v16h load_a16(const _Float16* A, int lda, int lane) {
  const int m = lane & 15, hv = (lane >> 4) & 1;
  const _Float16* p = A + m * lda + 8 * hv;
  const v8h lo = *(const v8h*)p;
  const v8h hi = *(const v8h*)(p + 16);
  v16h r;
#pragma unroll
  for (int i = 0; i < 8; ++i) { r[i] = lo[i]; r[8 + i] = hi[i]; }
  return r;
}

// Swizzled B storage: per 32(K) x 16(N) tile, lane l = (n | hv<<4) holds the
// 16 contiguous halfs B[32*kt + 8*hv + (e&7) + 16*(e>>3)][16*nt + n], e=0..15, at
// dst[(kt*ntiles + nt)*512 + l*16 + e].  Fragment load = 2x ds_load_b128.
DEV_INLINE v16h load_bsw(const _Float16* Bsw, int ntiles, int kt, int nt,
                         int lane) {
  const _Float16* p = Bsw + (((kt * ntiles + nt) << 9) + (lane << 4));
  const v8h lo = *(const v8h*)p;
  const v8h hi = *(const v8h*)(p + 8);
  v16h r;
#pragma unroll
  for (int i = 0; i < 8; ++i) { r[i] = lo[i]; r[8 + i] = hi[i]; }
  return r;
}

// Convert row-major f32 W[K x N] -> f16 B-fragment-swizzled LDS image.
DEV_INLINE void stage_bsw(_Float16* dst, const float* __restrict__ W, int K,
                          int N, int tid, int nthreads) {
  const int ntiles = N >> 4;
  const int total  = K * N;
  for (int i = tid; i < total; i += nthreads) {
    const int tile = i >> 9;
    const int rem  = i & 511;
    const int lane = rem >> 4, e = rem & 15;
    const int kt = tile / ntiles, nt = tile - kt * ntiles;
    const int n = lane & 15, hv = lane >> 4;
    const int row = kt * 32 + hv * 8 + (e & 7) + ((e >> 3) << 4);
    const int col = nt * 16 + n;
    dst[i] = (_Float16)W[row * N + col];
  }
}

DEV_INLINE v8f wmma32(v16h a, v16h b, v8f c) {
  return __builtin_amdgcn_wmma_f32_16x16x32_f16(false, a, false, b, (short)0, c,
                                                false, false);
}

// out[M x 64] = A[M x 64] @ W[64 x 64]
// mode 0: A(m,k) = inputs[b, t=k, f] with row r = b*256+f  (x_proj)
// mode 1: A(m,k) = A[row*64 + k]                            (xe_proj)
__global__ void proj_gemm_kernel(const float* __restrict__ A,
                                 const float* __restrict__ W,
                                 float* __restrict__ out, int mode) {
  extern __shared__ char smem[];
  _Float16* As = (_Float16*)smem;  // 16 x 64
  _Float16* Ws = As + 16 * 64;     // 64 x 64 swizzled
  const int tid = threadIdx.x;     // 128
  const int mt  = blockIdx.x;
  for (int i = tid; i < 16 * 64; i += 128) {
    const int m = i >> 6, k = i & 63;
    float v;
    if (mode == 0) {
      const int r = mt * 16 + m;
      const int b = r >> 8, f = r & 255;
      v = A[(size_t)b * (Tn * Fn) + (size_t)k * Fn + f];
    } else {
      v = A[(size_t)(mt * 16 + m) * 64 + k];
    }
    As[i] = (_Float16)v;
  }
  stage_bsw(Ws, W, 64, 64, tid, 128);
  __syncthreads();
  const int wave = tid >> 5, lane = tid & 31;
  v8f acc = {};
#pragma unroll
  for (int kt = 0; kt < 2; ++kt)
    acc = wmma32(load_a16(As + kt * 32, 64, lane),
                 load_bsw(Ws, 4, kt, wave, lane), acc);
  const int n = lane & 15, hv = (lane >> 4) & 1;
#pragma unroll
  for (int i = 0; i < 8; ++i)
    out[(size_t)(mt * 16 + 8 * hv + i) * 64 + wave * 16 + n] = acc[i];
}

// Persistent encoder: one WG per 16-batch tile, loops all T steps.
__global__ void encoder_kernel(
    const float* __restrict__ inputs, const float* __restrict__ enc_kernel,
    const float* __restrict__ enc_rec, const float* __restrict__ enc_bias,
    const float* __restrict__ attn_e_W, const float* __restrict__ attn_e_b,
    const float* __restrict__ attn_e_v, const float* __restrict__ attn_e_vb,
    const float* __restrict__ xproj, float* __restrict__ xenc) {
  extern __shared__ char smem[];
  char* p = smem;
  auto carve = [&](size_t bytes) { char* r = p; p += bytes; return r; };
  _Float16* Wh16 = (_Float16*)carve(4096 * 2);   // We_h 64x64 swz
  _Float16* Ws16 = (_Float16*)carve(4096 * 2);   // We_s 64x64 swz
  _Float16* ek16 = (_Float16*)carve(65536 * 2);  // enc_kernel 256x256 swz
  _Float16* er16 = (_Float16*)carve(16384 * 2);  // enc_rec 64x256 swz
  _Float16* h16  = (_Float16*)carve(1024 * 2);
  _Float16* s16  = (_Float16*)carve(1024 * 2);
  _Float16* xt16 = (_Float16*)carve(4096 * 2);   // x_tilde 16x256
  float* hS   = (float*)carve(1024 * 4);
  float* sS   = (float*)carve(1024 * 4);
  float* hs   = (float*)carve(1024 * 4);
  float* sc   = (float*)carve(4096 * 4);
  float* zS   = (float*)carve(4096 * 4);
  float* ve   = (float*)carve(64 * 4);
  float* be   = (float*)carve(64 * 4);
  float* encb = (float*)carve(256 * 4);
  float* red  = (float*)carve(256 * 4);

  const int tid = threadIdx.x;  // 256
  const int b0  = blockIdx.x * 16;
  stage_bsw(Wh16, attn_e_W, 64, 64, tid, 256);
  stage_bsw(Ws16, attn_e_W + 4096, 64, 64, tid, 256);
  stage_bsw(ek16, enc_kernel, 256, 256, tid, 256);
  stage_bsw(er16, enc_rec, 64, 256, tid, 256);
  if (tid < 64) { ve[tid] = attn_e_v[tid]; be[tid] = attn_e_b[tid]; }
  encb[tid] = enc_bias[tid];
  for (int i = tid; i < 1024; i += 256) { hS[i] = 0.f; sS[i] = 0.f; }
  const float vbv = attn_e_vb[0];
  __syncthreads();

  const int wave = tid >> 5, lane = tid & 31;
  const int bq = tid >> 4, jq = tid & 15;

  for (int t = 0; t < Tn; ++t) {
    for (int i = tid; i < 1024; i += 256) {
      h16[i] = (_Float16)hS[i];
      s16[i] = (_Float16)sS[i];
    }
    __syncthreads();
    // hs = h @ We_h + s @ We_s + b   (16x64, K=64)
    if (wave < 4) {
      v8f acc = {};
#pragma unroll
      for (int kt = 0; kt < 2; ++kt) {
        acc = wmma32(load_a16(h16 + kt * 32, 64, lane),
                     load_bsw(Wh16, 4, kt, wave, lane), acc);
        acc = wmma32(load_a16(s16 + kt * 32, 64, lane),
                     load_bsw(Ws16, 4, kt, wave, lane), acc);
      }
      const int n = lane & 15, hv = (lane >> 4) & 1;
#pragma unroll
      for (int i = 0; i < 8; ++i) {
        const int col = wave * 16 + n;
        hs[(8 * hv + i) * 64 + col] = acc[i] + be[col];
      }
    }
    __syncthreads();
    // score[b,f] = sum_u tanh(hs[b,u] + xproj[b,f,u]) * v[u]
    for (int i = tid; i < 16 * 256; i += 256) {
      const int b = i >> 8, f = i & 255;
      const float* xp = xproj + ((size_t)(b0 + b) * Fn + f) * Tn;
      if (b < 15) __builtin_prefetch(xp + (size_t)Fn * Tn, 0, 1);
      const float* hb = hs + b * 64;
      float a = 0.f;
#pragma unroll 4
      for (int u = 0; u < 64; ++u) a += d_tanh(hb[u] + xp[u]) * ve[u];
      sc[i] = a + vbv;
    }
    __syncthreads();
    // softmax over f per batch row; x_tilde = alpha * x_t  (-> f16)
    {
      float mx = -1e30f;
      for (int f = jq; f < 256; f += 16) mx = fmaxf(mx, sc[bq * 256 + f]);
      red[tid] = mx;
      __syncthreads();
      if (jq == 0) {
        float m2 = red[bq * 16];
        for (int q = 1; q < 16; ++q) m2 = fmaxf(m2, red[bq * 16 + q]);
        red[bq * 16] = m2;
      }
      __syncthreads();
      const float rmax = red[bq * 16];
      float ss = 0.f;
      for (int f = jq; f < 256; f += 16) ss += __expf(sc[bq * 256 + f] - rmax);
      __syncthreads();
      red[tid] = ss;
      __syncthreads();
      if (jq == 0) {
        float s2 = 0.f;
        for (int q = 0; q < 16; ++q) s2 += red[bq * 16 + q];
        red[bq * 16] = s2;
      }
      __syncthreads();
      const float inv  = 1.f / red[bq * 16];
      const float* xrw = inputs + (size_t)(b0 + bq) * Tn * Fn + (size_t)t * Fn;
      for (int f = jq; f < 256; f += 16) {
        const float alpha = __expf(sc[bq * 256 + f] - rmax) * inv;
        xt16[bq * 256 + f] = (_Float16)(alpha * xrw[f]);
      }
    }
    __syncthreads();
    // z = x_tilde @ enc_kernel + h @ enc_rec + b   (16x256)
    for (int nt = wave; nt < 16; nt += 8) {
      v8f acc = {};
#pragma unroll
      for (int kt = 0; kt < 8; ++kt)
        acc = wmma32(load_a16(xt16 + kt * 32, 256, lane),
                     load_bsw(ek16, 16, kt, nt, lane), acc);
#pragma unroll
      for (int kt = 0; kt < 2; ++kt)
        acc = wmma32(load_a16(h16 + kt * 32, 64, lane),
                     load_bsw(er16, 16, kt, nt, lane), acc);
      const int n = lane & 15, hv = (lane >> 4) & 1;
#pragma unroll
      for (int i = 0; i < 8; ++i) {
        const int col = nt * 16 + n;
        zS[(8 * hv + i) * 256 + col] = acc[i] + encb[col];
      }
    }
    __syncthreads();
    // LSTM gates, update h/s, emit h to x_encoded
    for (int i = tid; i < 1024; i += 256) {
      const int b = i >> 6, u = i & 63;
      const float zi = zS[b * 256 + u];
      const float zf = zS[b * 256 + 64 + u];
      const float zg = zS[b * 256 + 128 + u];
      const float zo = zS[b * 256 + 192 + u];
      const float cn = d_sigm(zf) * sS[i] + d_sigm(zi) * d_tanh(zg);
      const float hn = d_sigm(zo) * d_tanh(cn);
      sS[i] = cn;
      hS[i] = hn;
      xenc[((size_t)(b0 + b) * Tn + t) * Hn + u] = hn;
    }
    __syncthreads();
  }
}

// Persistent decoder: one WG per 16-batch tile.
__global__ void decoder_kernel(
    const float* __restrict__ inputs, const float* __restrict__ dec_kernel,
    const float* __restrict__ dec_rec, const float* __restrict__ dec_bias,
    const float* __restrict__ attn_d_W, const float* __restrict__ attn_d_b,
    const float* __restrict__ attn_d_v, const float* __restrict__ attn_d_vb,
    const float* __restrict__ fc_W, const float* __restrict__ fc_b,
    const float* __restrict__ xenc, const float* __restrict__ xeproj,
    float* __restrict__ dn_out, float* __restrict__ ctx_out) {
  extern __shared__ char smem[];
  char* p = smem;
  auto carve = [&](size_t bytes) { char* r = p; p += bytes; return r; };
  _Float16* Wdd16 = (_Float16*)carve(4096 * 2);   // swz
  _Float16* Wdc16 = (_Float16*)carve(4096 * 2);   // swz
  _Float16* fcW16 = (_Float16*)carve(4096 * 2);   // swz (fc_W rows 0..63)
  _Float16* dk16  = (_Float16*)carve(16384 * 2);  // swz 64x256
  _Float16* dr16  = (_Float16*)carve(16384 * 2);  // swz 64x256
  _Float16* xe16  = (_Float16*)carve(65536 * 2);  // x_encoded tile 16x64x64
  _Float16* d16   = (_Float16*)carve(1024 * 2);
  _Float16* c16   = (_Float16*)carve(1024 * 2);
  _Float16* ctx16 = (_Float16*)carve(1024 * 2);
  _Float16* yt16  = (_Float16*)carve(1024 * 2);
  float* dS   = (float*)carve(1024 * 4);
  float* cS   = (float*)carve(1024 * 4);
  float* dc   = (float*)carve(1024 * 4);
  float* sc   = (float*)carve(1024 * 4);
  float* ctx  = (float*)carve(1024 * 4);
  float* zS   = (float*)carve(4096 * 4);
  float* vd   = (float*)carve(64 * 4);
  float* bd   = (float*)carve(64 * 4);
  float* decb = (float*)carve(256 * 4);
  float* fcl  = (float*)carve(64 * 4);
  float* fcbS = (float*)carve(64 * 4);
  float* red  = (float*)carve(256 * 4);

  const int tid = threadIdx.x;  // 256
  const int b0  = blockIdx.x * 16;
  stage_bsw(Wdd16, attn_d_W, 64, 64, tid, 256);
  stage_bsw(Wdc16, attn_d_W + 4096, 64, 64, tid, 256);
  stage_bsw(fcW16, fc_W, 64, 64, tid, 256);
  stage_bsw(dk16, dec_kernel, 64, 256, tid, 256);
  stage_bsw(dr16, dec_rec, 64, 256, tid, 256);
  for (int i = tid; i < 65536; i += 256)
    xe16[i] = (_Float16)xenc[(size_t)b0 * (Tn * Hn) + i];
  if (tid < 64) {
    vd[tid]   = attn_d_v[tid];
    bd[tid]   = attn_d_b[tid];
    fcl[tid]  = fc_W[64 * 64 + tid];  // y_prev row of fc_W
    fcbS[tid] = fc_b[tid];
  }
  decb[tid] = dec_bias[tid];
  for (int i = tid; i < 1024; i += 256) { dS[i] = 0.f; cS[i] = 0.f; }
  const float vbv = attn_d_vb[0];
  __syncthreads();

  const int wave = tid >> 5, lane = tid & 31;
  const int bq = tid >> 4, jq = tid & 15;

  for (int t = 0; t < Tn; ++t) {
    for (int i = tid; i < 1024; i += 256) {
      d16[i] = (_Float16)dS[i];
      c16[i] = (_Float16)cS[i];
    }
    __syncthreads();
    // dc = d @ Wd_d + c @ Wd_c + b
    if (wave < 4) {
      v8f acc = {};
#pragma unroll
      for (int kt = 0; kt < 2; ++kt) {
        acc = wmma32(load_a16(d16 + kt * 32, 64, lane),
                     load_bsw(Wdd16, 4, kt, wave, lane), acc);
        acc = wmma32(load_a16(c16 + kt * 32, 64, lane),
                     load_bsw(Wdc16, 4, kt, wave, lane), acc);
      }
      const int n = lane & 15, hv = (lane >> 4) & 1;
#pragma unroll
      for (int i = 0; i < 8; ++i) {
        const int col = wave * 16 + n;
        dc[(8 * hv + i) * 64 + col] = acc[i] + bd[col];
      }
    }
    __syncthreads();
    // score[b,tt] = sum_u tanh(dc[b,u] + xeproj[b,tt,u]) * v[u]
    for (int i = tid; i < 1024; i += 256) {
      const int b = i >> 6, tt = i & 63;
      const float* xp = xeproj + ((size_t)(b0 + b) * Tn + tt) * Hn;
      const float* db = dc + b * 64;
      float a = 0.f;
#pragma unroll 4
      for (int u = 0; u < 64; ++u) a += d_tanh(db[u] + xp[u]) * vd[u];
      sc[i] = a + vbv;
    }
    __syncthreads();
    // softmax over T (64)
    {
      float mx = -1e30f;
      for (int f = jq; f < 64; f += 16) mx = fmaxf(mx, sc[bq * 64 + f]);
      red[tid] = mx;
      __syncthreads();
      if (jq == 0) {
        float m2 = red[bq * 16];
        for (int q = 1; q < 16; ++q) m2 = fmaxf(m2, red[bq * 16 + q]);
        red[bq * 16] = m2;
      }
      __syncthreads();
      const float rmax = red[bq * 16];
      float ss = 0.f;
      for (int f = jq; f < 64; f += 16) ss += __expf(sc[bq * 64 + f] - rmax);
      __syncthreads();
      red[tid] = ss;
      __syncthreads();
      if (jq == 0) {
        float s2 = 0.f;
        for (int q = 0; q < 16; ++q) s2 += red[bq * 16 + q];
        red[bq * 16] = s2;
      }
      __syncthreads();
      const float inv = 1.f / red[bq * 16];
      for (int f = jq; f < 64; f += 16)
        sc[bq * 64 + f] = __expf(sc[bq * 64 + f] - rmax) * inv;
    }
    __syncthreads();
    // context[b,h] = sum_t beta[b,t] * x_enc[b,t,h]
    for (int i = tid; i < 1024; i += 256) {
      const int b = i >> 6, hh = i & 63;
      float a = 0.f;
#pragma unroll 4
      for (int u = 0; u < 64; ++u)
        a += sc[b * 64 + u] * (float)xe16[b * 4096 + u * 64 + hh];
      ctx[i]   = a;
      ctx16[i] = (_Float16)a;
    }
    __syncthreads();
    // y_tilde = context @ fc_W[:64] + y_t * fc_W[64] + fc_b
    if (wave < 4) {
      v8f acc = {};
#pragma unroll
      for (int kt = 0; kt < 2; ++kt)
        acc = wmma32(load_a16(ctx16 + kt * 32, 64, lane),
                     load_bsw(fcW16, 4, kt, wave, lane), acc);
      const int n = lane & 15, hv = (lane >> 4) & 1;
#pragma unroll
      for (int i = 0; i < 8; ++i) {
        const int m = 8 * hv + i, col = wave * 16 + n;
        const float yv =
            inputs[(size_t)(b0 + m) * Tn * Fn + (size_t)t * Fn + (Fn - 1)];
        yt16[m * 64 + col] = (_Float16)(acc[i] + yv * fcl[col] + fcbS[col]);
      }
    }
    __syncthreads();
    // z = y_tilde @ dec_kernel + d @ dec_rec + b
    for (int nt = wave; nt < 16; nt += 8) {
      v8f acc = {};
#pragma unroll
      for (int kt = 0; kt < 2; ++kt) {
        acc = wmma32(load_a16(yt16 + kt * 32, 64, lane),
                     load_bsw(dk16, 16, kt, nt, lane), acc);
        acc = wmma32(load_a16(d16 + kt * 32, 64, lane),
                     load_bsw(dr16, 16, kt, nt, lane), acc);
      }
      const int n = lane & 15, hv = (lane >> 4) & 1;
#pragma unroll
      for (int i = 0; i < 8; ++i) {
        const int col = nt * 16 + n;
        zS[(8 * hv + i) * 256 + col] = acc[i] + decb[col];
      }
    }
    __syncthreads();
    for (int i = tid; i < 1024; i += 256) {
      const int b = i >> 6, u = i & 63;
      const float zi = zS[b * 256 + u];
      const float zf = zS[b * 256 + 64 + u];
      const float zg = zS[b * 256 + 128 + u];
      const float zo = zS[b * 256 + 192 + u];
      const float cn = d_sigm(zf) * cS[i] + d_sigm(zi) * d_tanh(zg);
      dS[i] = d_sigm(zo) * d_tanh(cn);
      cS[i] = cn;
    }
    __syncthreads();
  }
  for (int i = tid; i < 1024; i += 256) {
    dn_out[(size_t)b0 * Hn + i]  = dS[i];
    ctx_out[(size_t)b0 * Hn + i] = ctx[i];
  }
}

// y_pred = [d_n, context] @ fcf_W + fcf_b   (256x256, K=128)
__global__ void final_kernel(const float* __restrict__ dn,
                             const float* __restrict__ ctx,
                             const float* __restrict__ fcf_W,
                             const float* __restrict__ fcf_b,
                             float* __restrict__ out) {
  extern __shared__ char smem[];
  _Float16* As = (_Float16*)smem;  // 16 x 128
  _Float16* Ws = As + 16 * 128;    // 128 x 256 swizzled
  const int tid = threadIdx.x;     // 128
  const int b0  = blockIdx.x * 16;
  for (int i = tid; i < 16 * 128; i += 128) {
    const int m = i >> 7, k = i & 127;
    const float v = (k < 64) ? dn[(size_t)(b0 + m) * 64 + k]
                             : ctx[(size_t)(b0 + m) * 64 + (k - 64)];
    As[i] = (_Float16)v;
  }
  stage_bsw(Ws, fcf_W, 128, 256, tid, 128);
  __syncthreads();
  const int wave = tid >> 5, lane = tid & 31;
  for (int nt = wave; nt < 16; nt += 4) {
    v8f acc = {};
#pragma unroll
    for (int kt = 0; kt < 4; ++kt)
      acc = wmma32(load_a16(As + kt * 32, 128, lane),
                   load_bsw(Ws, 16, kt, nt, lane), acc);
    const int n = lane & 15, hv = (lane >> 4) & 1;
#pragma unroll
    for (int i = 0; i < 8; ++i) {
      const int col = nt * 16 + n;
      out[(size_t)(b0 + 8 * hv + i) * 256 + col] = acc[i] + fcf_b[col];
    }
  }
}

extern "C" void kernel_launch(void* const* d_in, const int* in_sizes, int n_in,
                              void* d_out, int out_size, void* d_ws,
                              size_t ws_size, hipStream_t stream) {
  (void)in_sizes; (void)n_in; (void)out_size; (void)ws_size;
  const float* inputs     = (const float*)d_in[0];
  const float* enc_kernel = (const float*)d_in[1];
  const float* enc_rec    = (const float*)d_in[2];
  const float* enc_bias   = (const float*)d_in[3];
  const float* dec_kernel = (const float*)d_in[4];
  const float* dec_rec    = (const float*)d_in[5];
  const float* dec_bias   = (const float*)d_in[6];
  const float* attn_e_W   = (const float*)d_in[7];
  const float* attn_e_b   = (const float*)d_in[8];
  const float* attn_e_v   = (const float*)d_in[9];
  const float* attn_e_vb  = (const float*)d_in[10];
  const float* attn_d_W   = (const float*)d_in[11];
  const float* attn_d_b   = (const float*)d_in[12];
  const float* attn_d_v   = (const float*)d_in[13];
  const float* attn_d_vb  = (const float*)d_in[14];
  const float* fc_W       = (const float*)d_in[15];
  const float* fc_b       = (const float*)d_in[16];
  const float* fcf_W      = (const float*)d_in[17];
  const float* fcf_b      = (const float*)d_in[18];
  float* out = (float*)d_out;

  float* x_proj  = (float*)d_ws;                    // B*F*T f32 (16 MB)
  float* x_enc   = x_proj + (size_t)Bn * Fn * Tn;   // B*T*H  (4 MB)
  float* xe_proj = x_enc + (size_t)Bn * Tn * Hn;    // B*T*H  (4 MB)
  float* dn      = xe_proj + (size_t)Bn * Tn * Hn;  // B*H
  float* ctxf    = dn + (size_t)Bn * Hn;            // B*H

  constexpr size_t PROJ_LDS = (16 * 64 + 64 * 64) * sizeof(_Float16);
  constexpr size_t ENC_LDS =
      (size_t)(4096 + 4096 + 65536 + 16384 + 1024 + 1024 + 4096) *
          sizeof(_Float16) +
      (size_t)(1024 + 1024 + 1024 + 4096 + 4096 + 64 + 64 + 256 + 256) *
          sizeof(float);  // ~235 KB (fits 320 KB CDNA5 LDS)
  constexpr size_t DEC_LDS =
      (size_t)(4096 * 3 + 16384 * 2 + 65536 + 1024 * 4) * sizeof(_Float16) +
      (size_t)(1024 * 5 + 4096 + 64 * 2 + 256 + 64 * 2 + 256) *
          sizeof(float);  // ~263 KB
  constexpr size_t FIN_LDS = (16 * 128 + 128 * 256) * sizeof(_Float16);

  // x_proj[b,f,:] = inputs[b,:,f] @ We_x  (We_x = attn_e_W rows 128..191)
  proj_gemm_kernel<<<(Bn * Fn) / 16, 128, PROJ_LDS, stream>>>(
      inputs, attn_e_W + 2 * Hn * Tn, x_proj, 0);
  encoder_kernel<<<Bn / 16, 256, ENC_LDS, stream>>>(
      inputs, enc_kernel, enc_rec, enc_bias, attn_e_W, attn_e_b, attn_e_v,
      attn_e_vb, x_proj, x_enc);
  // xe_proj = x_encoded @ Wd_x  (Wd_x = attn_d_W rows 128..191)
  proj_gemm_kernel<<<(Bn * Tn) / 16, 128, PROJ_LDS, stream>>>(
      x_enc, attn_d_W + 2 * Hn * Hn, xe_proj, 1);
  decoder_kernel<<<Bn / 16, 256, DEC_LDS, stream>>>(
      inputs, dec_kernel, dec_rec, dec_bias, attn_d_W, attn_d_b, attn_d_v,
      attn_d_vb, fc_W, fc_b, x_enc, xe_proj, dn, ctxf);
  final_kernel<<<Bn / 16, 128, FIN_LDS, stream>>>(dn, ctxf, fcf_W, fcf_b, out);
}
